// Monet_8770323219141
// MI455X (gfx1250) — compile-verified, matrix operand order checked by reading, not run on hardware
//
#include <hip/hip_runtime.h>
#include <hip/hip_bf16.h>

typedef unsigned short u16;
typedef unsigned int   u32;
typedef unsigned long long u64;

typedef __attribute__((ext_vector_type(16))) __bf16 bf16x16;
typedef __attribute__((ext_vector_type(8)))  float  floatx8;
typedef __attribute__((ext_vector_type(4)))  unsigned int uint32x4;
typedef __attribute__((ext_vector_type(8)))  int int32x8;
typedef __attribute__((ext_vector_type(4)))  int int32x4;

union FragT { bf16x16 v; u32 u[8]; };
union AccT  { floatx8 v; float f[8]; };

enum : int {
  Ntok = 16384,
  Dh   = 1024,
  Hh   = 6,
  Ee   = 512,
  Mm   = 6,
  Ff   = 3072,            // E*M
  NC1  = 4 * Ff,          // 12288 = stage-1 output columns (gz1|gz2|z1|z2)
  KC2  = 2 * Ff + Ee,     // 6656  = stage-2 K (A11|A12|g1s)
};

__device__ __forceinline__ u16 f2bf(float f) {
  u32 u = __float_as_uint(f);
  u32 r = u + 0x7FFFu + ((u >> 16) & 1u);   // round-to-nearest-even
  return (u16)(r >> 16);
}
__device__ __forceinline__ float bf2f(u16 h) {
  return __uint_as_float(((u32)h) << 16);
}

// generic -> LDS(AS3) byte offset
__device__ __forceinline__ u32 lds_off(const void* p) {
#if defined(__HIP_DEVICE_COMPILE__)
  return (u32)(size_t)(const __attribute__((address_space(3))) void*)p;
#else
  return 0;
#endif
}

// ---------------------------------------------------------------------------
// TDM: async-load a 2D bf16 tile [tile_d1 rows x tile_d0 elems] from global
// (row stride = stride_elems) into LDS at lds_byte, padding each 64B row by
// one DWORD (matches LDS row pitch of 34 halves). Tracked by TENSORcnt.
// ---------------------------------------------------------------------------
__device__ __forceinline__ void tdm_load_tile_bf16(
    u32 lds_byte, const u16* gsrc, u32 tile_d0, u32 tile_d1, u32 stride_elems) {
  u64 ga = (u64)(size_t)gsrc;
  uint32x4 g0;
  g0[0] = 1u;                                           // count=1, no gather
  g0[1] = lds_byte;                                     // lds_addr
  g0[2] = (u32)ga;                                      // global_addr[31:0]
  g0[3] = ((u32)(ga >> 32) & 0x01FFFFFFu) | (2u << 30); // addr[56:32] | type=2

  int32x8 g1;
  // data_size=1 (2B), pad_enable, pad_interval=3 (16 DWORDs), pad_amount=0 (1 DWORD)
  g1[0] = (int)((1u << 16) | (1u << 20) | (3u << 22));
  g1[1] = (int)((tile_d0 & 0xFFFFu) << 16);             // tensor_dim0[15:0]
  g1[2] = (int)(((tile_d0 >> 16) & 0xFFFFu) | ((tile_d1 & 0xFFFFu) << 16));
  g1[3] = (int)(((tile_d1 >> 16) & 0xFFFFu) | ((tile_d0 & 0xFFFFu) << 16)); // tile_dim0
  g1[4] = (int)(tile_d1 & 0xFFFFu);                     // tile_dim1, tile_dim2=0
  g1[5] = (int)stride_elems;                            // tensor_dim0_stride[31:0]
  g1[6] = 0;
  g1[7] = 0;

  int32x4 z4 = {0, 0, 0, 0};
#if __clang_major__ >= 23
  int32x8 z8 = {0, 0, 0, 0, 0, 0, 0, 0};
  __builtin_amdgcn_tensor_load_to_lds(g0, g1, z4, z4, z8, 0);
#else
  __builtin_amdgcn_tensor_load_to_lds(g0, g1, z4, z4, 0);
#endif
}

// ---------------------------------------------------------------------------
// Pack / convert kernels
// ---------------------------------------------------------------------------
__global__ __launch_bounds__(256) void k_pack_x(const float* __restrict__ x,
                                                u16* __restrict__ xb) {
  int i = blockIdx.x * 256 + threadIdx.x;
  if (i < Ntok * Dh) xb[i] = f2bf(x[i]);
}

__global__ __launch_bounds__(256) void k_pack_w(const float* __restrict__ w1,
                                                const float* __restrict__ w2,
                                                const float* __restrict__ u1,
                                                const float* __restrict__ u2,
                                                u16* __restrict__ W) {
  int i = blockIdx.x * 256 + threadIdx.x;
  if (i >= NC1 * Dh) return;
  int row = i >> 10;                   // /1024
  float v;
  if      (row < 1 * Ff) v = w1[i];
  else if (row < 2 * Ff) v = w2[i - 1 * Ff * Dh];
  else if (row < 3 * Ff) v = u1[i - 2 * Ff * Dh];
  else                   v = u2[i - 3 * Ff * Dh];
  W[i] = f2bf(v);
}

__global__ __launch_bounds__(256) void k_pack_bias(const float* __restrict__ u1b,
                                                   const float* __restrict__ u2b,
                                                   float* __restrict__ bias) {
  int i = blockIdx.x * 256 + threadIdx.x;
  if (i >= NC1) return;
  float v = 0.0f;
  if (i >= 3 * Ff)      v = u2b[i - 3 * Ff];
  else if (i >= 2 * Ff) v = u1b[i - 2 * Ff];
  bias[i] = v;
}

// B_cat[n][k] (n<512, k<6656) = [vA | vB | bmat^T]
__global__ __launch_bounds__(256) void k_pack_bcat(const float* __restrict__ vA,
                                                   const float* __restrict__ vB,
                                                   const float* __restrict__ bmat,
                                                   u16* __restrict__ B) {
  int i = blockIdx.x * 256 + threadIdx.x;
  if (i >= Ee * KC2) return;
  int n = i / KC2;
  int k = i - n * KC2;
  float v;
  if      (k < Ff)     v = vA[(size_t)n * Ff + k];
  else if (k < 2 * Ff) v = vB[(size_t)n * Ff + (k - Ff)];
  else                 v = bmat[(size_t)(k - 2 * Ff) * Ee + n];  // b1[e][n]
  B[i] = f2bf(v);
}

// ---------------------------------------------------------------------------
// bf16 WMMA GEMM: C[m, col_off+n] = sum_k A[m,k] * Bt[n,k] + bias[n]
// Block tile 128x128, K-step 32, TDM-staged double-buffered LDS.
// grid = (Ncols/128, Mrows/128), block = 256 (8 waves of 32x64; 8 WMMA/step)
// ---------------------------------------------------------------------------
__global__ __launch_bounds__(256) void wmma_gemm_bf16(
    const u16* __restrict__ A,      // [Mrows, Kdim] bf16 row-major
    const u16* __restrict__ Bt,     // [Ncols, Kdim] bf16 row-major (B^T)
    const float* __restrict__ bias, // [Ncols] (never null)
    float* __restrict__ C,          // fp32, row stride ldc
    int Kdim, int ldc, int col_off) {
  __shared__ u16 As[2][128][34];    // 17 DWORD pitch (16 data + 1 pad)
  __shared__ u16 Bs[2][128][34];

  const int t       = threadIdx.x;
  const int rowBase = blockIdx.y * 128;
  const int colBase = blockIdx.x * 128;

  const int w    = t >> 5;
  const int lane = t & 31;
  const int wm   = (w & 3) << 5;    // 0,32,64,96  (row group)
  const int wn   = (w >> 2) << 6;   // 0,64        (col group)
  const int lrow = lane & 15;
  const int lh   = lane >> 4;

  AccT acc[2][4];
#pragma unroll
  for (int i = 0; i < 2; ++i)
#pragma unroll
    for (int j = 0; j < 4; ++j)
#pragma unroll
      for (int r = 0; r < 8; ++r) acc[i][j].f[r] = 0.0f;

  const u16* aTile = A  + (size_t)rowBase * Kdim;
  const u16* bTile = Bt + (size_t)colBase * Kdim;
  const u32 ldsA[2] = {lds_off(&As[0][0][0]), lds_off(&As[1][0][0])};
  const u32 ldsB[2] = {lds_off(&Bs[0][0][0]), lds_off(&Bs[1][0][0])};

  // prologue: TDM tile 0 -> buffer 0
  if (w == 0) {
    tdm_load_tile_bf16(ldsA[0], aTile, 32u, 128u, (u32)Kdim);
    tdm_load_tile_bf16(ldsB[0], bTile, 32u, 128u, (u32)Kdim);
    __builtin_amdgcn_s_wait_tensorcnt(0);
  }
  __syncthreads();

  const int nk = Kdim >> 5;
  for (int kb = 0; kb < nk; ++kb) {
    const int cur   = kb & 1;
    const bool more = (kb + 1) < nk;
    if (more && w == 0) {
      tdm_load_tile_bf16(ldsA[cur ^ 1], aTile + (size_t)(kb + 1) * 32,
                         32u, 128u, (u32)Kdim);
      tdm_load_tile_bf16(ldsB[cur ^ 1], bTile + (size_t)(kb + 1) * 32,
                         32u, 128u, (u32)Kdim);
    }

    // fragments per ISA 16-bit A/B VGPR layout (K-pairs, lane halves)
    FragT fa[2], fb[4];
#pragma unroll
    for (int r = 0; r < 8; ++r) {
      int k0 = 2 * r + 8 * lh + ((r >= 4) ? 8 : 0);
      fa[0].u[r] = *(const u32*)&As[cur][wm + lrow][k0];
      fa[1].u[r] = *(const u32*)&As[cur][wm + 16 + lrow][k0];
#pragma unroll
      for (int j = 0; j < 4; ++j)
        fb[j].u[r] = *(const u32*)&Bs[cur][wn + j * 16 + lrow][k0];
    }
#pragma unroll
    for (int i = 0; i < 2; ++i)
#pragma unroll
      for (int j = 0; j < 4; ++j)
        acc[i][j].v = __builtin_amdgcn_wmma_f32_16x16x32_bf16(
            false, fa[i].v, false, fb[j].v, (short)0, acc[i][j].v, false, false);

    if (more) {
      if (w == 0) __builtin_amdgcn_s_wait_tensorcnt(0);
      // publishes next buffer AND guarantees everyone is done reading `cur`
      // before next iteration's TDM overwrites it
      __syncthreads();
    }
  }

  // epilogue: C layout = 8 VGPRs, row = r + 8*(lane/16), col = lane%16
#pragma unroll
  for (int i = 0; i < 2; ++i) {
#pragma unroll
    for (int j = 0; j < 4; ++j) {
      int gcol = colBase + wn + j * 16 + lrow;
      float bv = bias[gcol];
#pragma unroll
      for (int r = 0; r < 8; ++r) {
        int grow = rowBase + wm + i * 16 + r + 8 * lh;
        C[(size_t)grow * ldc + col_off + gcol] = acc[i][j].f[r] + bv;
      }
    }
  }
}

// ---------------------------------------------------------------------------
// Gating + mixing: one block (256 threads) per token.
// Reads out1 = [gz1|gz2|z1|z2] (fp32), writes Acat1/Acat2 (bf16, K=6656):
//   Acat1 = [A11 = x1*g1s | A12 = (g1^T t2) | g1s]
//   Acat2 = [A21 = (g2^T t1) | A22 = x2*g2s | g2s]
// ---------------------------------------------------------------------------
__global__ __launch_bounds__(256) void k_gate_mix(
    const float* __restrict__ out1,
    const float* __restrict__ ln1w, const float* __restrict__ ln2w,
    u16* __restrict__ Acat1, u16* __restrict__ Acat2) {
  const int tok = blockIdx.x;
  const int t   = threadIdx.x;

  __shared__ float sGz[Hh][Ee];       // 12 KB  current gate logits
  __shared__ float sG[2][Hh][Ee];     // 24 KB  softmax gates g1,g2
  __shared__ u16   sXh[2][Ee][Mm];    // 12 KB  relu^2(z) in bf16
  __shared__ float sGs[2][Ee];        //  4 KB  head-summed gates
  __shared__ float sWork[Ee];         //  2 KB
  __shared__ float sRedV[256];
  __shared__ int   sRedI[256];
  __shared__ float sT[2][Hh][Mm];     // t1, t2
  __shared__ float sTh, sMax, sSum;

  const float* base = out1 + (size_t)tok * NC1;

  for (int gid = 0; gid < 2; ++gid) {
    const float* lnw = gid ? ln2w : ln1w;

    for (int i = t; i < Ff; i += 256) sGz[i >> 9][i & 511] = base[gid * Ff + i];
    for (int i = t; i < Ff; i += 256) {
      float z = base[2 * Ff + gid * Ff + i];
      float r = fmaxf(z, 0.0f);
      sXh[gid][i / Mm][i % Mm] = f2bf(r * r);
    }
    __syncthreads();

    for (int h = 0; h < Hh; ++h) {
      const float wh = lnw[h];

      // gn for this head into sWork (LayerNorm over head axis)
      for (int e = t; e < Ee; e += 256) {
        float v[Hh], mu = 0.0f, s2 = 0.0f;
#pragma unroll
        for (int q = 0; q < Hh; ++q) { v[q] = sGz[q][e]; mu += v[q]; }
        mu *= (1.0f / Hh);
#pragma unroll
        for (int q = 0; q < Hh; ++q) { float d = v[q] - mu; s2 += d * d; }
        s2 *= (1.0f / Hh);
        sWork[e] = (v[h] - mu) * rsqrtf(s2 + 1e-5f) * wh;
      }
      __syncthreads();

      // exact top-8 threshold via 8 argmax reductions
      for (int it = 0; it < 8; ++it) {
        float bv = sWork[t];  int bi = t;
        float ov = sWork[t + 256];
        if (ov > bv) { bv = ov; bi = t + 256; }
        sRedV[t] = bv; sRedI[t] = bi;
        __syncthreads();
        for (int s = 128; s > 0; s >>= 1) {
          if (t < s && sRedV[t + s] > sRedV[t]) {
            sRedV[t] = sRedV[t + s]; sRedI[t] = sRedI[t + s];
          }
          __syncthreads();
        }
        if (t == 0) { sTh = sRedV[0]; sWork[sRedI[0]] = -1e30f; }
        __syncthreads();
      }
      const float thresh = sTh;

      // masked logits (recompute gn; softmax uses RAW gz as logits)
      for (int e = t; e < Ee; e += 256) {
        float v[Hh], mu = 0.0f, s2 = 0.0f;
#pragma unroll
        for (int q = 0; q < Hh; ++q) { v[q] = sGz[q][e]; mu += v[q]; }
        mu *= (1.0f / Hh);
#pragma unroll
        for (int q = 0; q < Hh; ++q) { float d = v[q] - mu; s2 += d * d; }
        s2 *= (1.0f / Hh);
        float gn = (v[h] - mu) * rsqrtf(s2 + 1e-5f) * wh;
        sWork[e] = (gn >= thresh) ? v[h] : -1e10f;
      }
      __syncthreads();

      // softmax over e
      sRedV[t] = fmaxf(sWork[t], sWork[t + 256]);
      __syncthreads();
      for (int s = 128; s > 0; s >>= 1) {
        if (t < s) sRedV[t] = fmaxf(sRedV[t], sRedV[t + s]);
        __syncthreads();
      }
      if (t == 0) sMax = sRedV[0];
      __syncthreads();
      const float mx = sMax;

      sRedV[t] = __expf(sWork[t] - mx) + __expf(sWork[t + 256] - mx);
      __syncthreads();
      for (int s = 128; s > 0; s >>= 1) {
        if (t < s) sRedV[t] += sRedV[t + s];
        __syncthreads();
      }
      if (t == 0) sSum = sRedV[0];
      __syncthreads();
      const float inv = 1.0f / sSum;

      for (int e = t; e < Ee; e += 256)
        sG[gid][h][e] = __expf(sWork[e] - mx) * inv;
      __syncthreads();
    }
  }

  // head sums g1s/g2s
  for (int e = t; e < Ee; e += 256) {
    float a = 0.0f, b = 0.0f;
#pragma unroll
    for (int q = 0; q < Hh; ++q) { a += sG[0][q][e]; b += sG[1][q][e]; }
    sGs[0][e] = a; sGs[1][e] = b;
  }
  if (t < 2 * Hh * Mm) ((float*)sT)[t] = 0.0f;
  __syncthreads();

  // t1[h][m] = sum_e x1[e][m] g1[h][e] ; t2 likewise with x2,g2
  for (int pass = 0; pass < 2; ++pass) {
    float accm[Hh * Mm];
#pragma unroll
    for (int q = 0; q < Hh * Mm; ++q) accm[q] = 0.0f;
    for (int j = t; j < Ee; j += 256) {
      float xv[Mm];
#pragma unroll
      for (int m = 0; m < Mm; ++m) xv[m] = bf2f(sXh[pass][j][m]);
#pragma unroll
      for (int h = 0; h < Hh; ++h) {
        float g = sG[pass][h][j];
#pragma unroll
        for (int m = 0; m < Mm; ++m) accm[h * Mm + m] += g * xv[m];
      }
    }
#pragma unroll
    for (int q = 0; q < Hh * Mm; ++q)
      atomicAdd(&((float*)sT)[pass * Hh * Mm + q], accm[q]);
  }
  __syncthreads();

  // emit stage-2 A matrices (bf16)
  const size_t ob = (size_t)tok * KC2;
  for (int e = t; e < Ee; e += 256) {
    const float gs0 = sGs[0][e], gs1 = sGs[1][e];
#pragma unroll
    for (int m = 0; m < Mm; ++m) {
      float x1v = bf2f(sXh[0][e][m]);
      float x2v = bf2f(sXh[1][e][m]);
      float a12 = 0.0f, a21 = 0.0f;
#pragma unroll
      for (int h = 0; h < Hh; ++h) {
        a12 += sT[1][h][m] * sG[0][h][e];   // t2 with g1
        a21 += sT[0][h][m] * sG[1][h][e];   // t1 with g2
      }
      Acat1[ob + e * Mm + m]      = f2bf(x1v * gs0);   // A11
      Acat1[ob + Ff + e * Mm + m] = f2bf(a12);         // A12
      Acat2[ob + e * Mm + m]      = f2bf(a21);         // A21
      Acat2[ob + Ff + e * Mm + m] = f2bf(x2v * gs1);   // A22
    }
    Acat1[ob + 2 * Ff + e] = f2bf(gs0);                // g1s (pairs with b1)
    Acat2[ob + 2 * Ff + e] = f2bf(gs1);                // g2s (pairs with b2)
  }
}

// ---------------------------------------------------------------------------
extern "C" void kernel_launch(void* const* d_in, const int* in_sizes, int n_in,
                              void* d_out, int out_size, void* d_ws, size_t ws_size,
                              hipStream_t stream) {
  const float* x   = (const float*)d_in[0];
  const float* w1  = (const float*)d_in[1];
  const float* w2  = (const float*)d_in[2];
  const float* ln1 = (const float*)d_in[3];
  const float* ln2 = (const float*)d_in[4];
  const float* u1w = (const float*)d_in[5];
  const float* u1b = (const float*)d_in[6];
  const float* u2w = (const float*)d_in[7];
  const float* u2b = (const float*)d_in[8];
  const float* v11 = (const float*)d_in[9];
  const float* v12 = (const float*)d_in[10];
  const float* v21 = (const float*)d_in[11];
  const float* v22 = (const float*)d_in[12];
  const float* b1  = (const float*)d_in[13];
  const float* b2  = (const float*)d_in[14];
  float* out = (float*)d_out;

  // linear workspace carve-out (256B aligned)
  char* p = (char*)d_ws;
  size_t off = 0;
  auto take = [&](size_t bytes) {
    void* r = p + off;
    off = (off + bytes + 255) & ~(size_t)255;
    return r;
  };
  u16*   xb      = (u16*)  take((size_t)Ntok * Dh * 2);        //  32 MB
  u16*   Wcat    = (u16*)  take((size_t)NC1 * Dh * 2);         //  24 MB
  float* biascat = (float*)take((size_t)NC1 * 4);              //  48 KB (first 6144 = 0)
  float* out1    = (float*)take((size_t)Ntok * NC1 * 4);       // 768 MB
  u16*   Acat1   = (u16*)  take((size_t)Ntok * KC2 * 2);       // 208 MB
  u16*   Acat2   = (u16*)  take((size_t)Ntok * KC2 * 2);       // 208 MB
  u16*   Bcat1   = (u16*)  take((size_t)Ee * KC2 * 2);         // 6.5 MB
  u16*   Bcat2   = (u16*)  take((size_t)Ee * KC2 * 2);         // 6.5 MB

  // pack / convert
  k_pack_x   <<<(Ntok * Dh + 255) / 256, 256, 0, stream>>>(x, xb);
  k_pack_w   <<<(NC1 * Dh + 255) / 256, 256, 0, stream>>>(w1, w2, u1w, u2w, Wcat);
  k_pack_bias<<<(NC1 + 255) / 256, 256, 0, stream>>>(u1b, u2b, biascat);
  k_pack_bcat<<<(Ee * KC2 + 255) / 256, 256, 0, stream>>>(v11, v12, b1, Bcat1);
  k_pack_bcat<<<(Ee * KC2 + 255) / 256, 256, 0, stream>>>(v21, v22, b2, Bcat2);

  // stage 1: [gz1|gz2|z1|z2] = x @ Wcat^T (+bias)   (16384x1024)x(1024x12288)
  wmma_gemm_bf16<<<dim3(NC1 / 128, Ntok / 128), 256, 0, stream>>>(
      xb, Wcat, biascat, out1, Dh, NC1, 0);

  // gating + mixing: build Acat1/Acat2
  k_gate_mix<<<Ntok, 256, 0, stream>>>(out1, ln1, ln2, Acat1, Acat2);

  // stage 2: out[:, :512]  = Acat1 @ Bcat1^T ; out[:, 512:] = Acat2 @ Bcat2^T
  // (bias = biascat[0..511] which is all zeros -> branch-free epilogue)
  wmma_gemm_bf16<<<dim3(Ee / 128, Ntok / 128), 256, 0, stream>>>(
      Acat1, Bcat1, biascat, out, KC2, 2 * Ee, 0);
  wmma_gemm_bf16<<<dim3(Ee / 128, Ntok / 128), 256, 0, stream>>>(
      Acat2, Bcat2, biascat, out, KC2, 2 * Ee, Ee);
}
